// Qwen3_5VisionAttention_16312285791067
// MI455X (gfx1250) — compile-verified, matrix operand order checked by reading, not run on hardware
//
#include <hip/hip_runtime.h>
#include <hip/hip_bf16.h>

// ---------- CDNA5 WMMA plumbing ----------
typedef __attribute__((ext_vector_type(16))) __bf16 v16bf;
typedef __attribute__((ext_vector_type(8)))  float  v8f;

struct __align__(16) U128 { unsigned int x, y, z, w; };
union BV { v16bf v; U128 u[2]; };

__device__ __forceinline__ v8f wmma_bf16(v16bf a, v16bf b, v8f c) {
  // (neg_a, A, neg_b, B, c_mod, C, reuse_a, reuse_b)
  return __builtin_amdgcn_wmma_f32_16x16x32_bf16(false, a, false, b, (short)0, c,
                                                 false, false);
}

__device__ __forceinline__ v8f v8f_zero() {
  v8f z;
#pragma unroll
  for (int i = 0; i < 8; ++i) z[i] = 0.0f;
  return z;
}

__device__ __forceinline__ unsigned short f2bf(float f) {
  unsigned int u = __float_as_uint(f);
  u += 0x7FFFu + ((u >> 16) & 1u);   // round-to-nearest-even
  return (unsigned short)(u >> 16);
}

// ---------- CDNA5 async global->LDS copy (ASYNCcnt path) ----------
__device__ __forceinline__ void async_copy_b128(unsigned lds_off, const void* gptr) {
  // per-lane: LDS[lds_off .. +15] = MEM[gptr .. +15], tracked by ASYNCcnt
  asm volatile("global_load_async_to_lds_b128 %0, %1, off"
               :: "v"(lds_off), "v"(gptr)
               : "memory");
}
__device__ __forceinline__ void wait_asynccnt0() {
  asm volatile("s_wait_asynccnt 0x0" ::: "memory");
}
__device__ __forceinline__ unsigned lds_addr32(const void* p) {
  // generic LDS pointer: low 32 bits are the LDS byte address (aperture truncation)
  return (unsigned)(unsigned long long)p;
}

// ---------- problem constants ----------
#define BB 4
#define SS 1024
#define DD 1280
#define HH 16
#define HD 80
#define HDP 96          // HD padded to 3*32 for WMMA K loop
#define BS (BB*SS)      // 4096
#define D3 (3*DD)       // 3840

// ---------- kernel 0: fp32 -> bf16 ----------
__global__ __launch_bounds__(256) void cvt_f32_bf16(const float* __restrict__ in,
                                                    unsigned short* __restrict__ out,
                                                    int n) {
  int i = blockIdx.x * 256 + threadIdx.x;
  if (i < n) out[i] = f2bf(in[i]);
}

// ---------- kernel 1/4: C[M,N] = A[M,K] * W[N,K]^T + bias, bf16 WMMA ----------
// 128x128 block tile, 8 waves as 2(m) x 4(n); each wave: 64x32 = acc[4][2].
// Double-buffered LDS staged with global_load_async_to_lds_b128.
#define BM 128
#define BN 128
#define BK 32
#define LDT 40   // padded LDS row stride (halfwords): 80B = 20 banks -> conflict free

__global__ __launch_bounds__(256) void gemm_bf16_bias(
    const unsigned short* __restrict__ A,   // [M,K] bf16
    const unsigned short* __restrict__ W,   // [N,K] bf16 (row n is K-contiguous)
    const float* __restrict__ bias,         // [N]
    float* __restrict__ C,                  // [M,N] fp32
    int M, int N, int K) {
  __shared__ __align__(16) unsigned short As[2][BM][LDT];
  __shared__ __align__(16) unsigned short Bs[2][BN][LDT];

  const int tid  = threadIdx.x;
  const int m0   = blockIdx.y * BM;
  const int n0   = blockIdx.x * BN;
  const int w    = tid >> 5;
  const int lane = tid & 31;
  const int r    = lane & 15;
  const int half = lane >> 4;
  const int wm   = (w & 1) * 64;   // wave m sub-tile (4 x 16 rows)
  const int wn   = (w >> 1) * 32;  // wave n sub-tile (2 x 16 cols)

  v8f acc[4][2];
#pragma unroll
  for (int i = 0; i < 4; ++i)
#pragma unroll
    for (int j = 0; j < 2; ++j) acc[i][j] = v8f_zero();

  // staging assignment: 256 threads, each copies 32B of A and 32B of B per k-step
  const int srow = tid >> 1, sseg = (tid & 1) * 16;
  const unsigned short* gA = A + (size_t)(m0 + srow) * K + sseg;
  const unsigned short* gB = W + (size_t)(n0 + srow) * K + sseg;

  // prologue: fill buffer 0
  async_copy_b128(lds_addr32(&As[0][srow][sseg]),     gA);
  async_copy_b128(lds_addr32(&As[0][srow][sseg + 8]), gA + 8);
  async_copy_b128(lds_addr32(&Bs[0][srow][sseg]),     gB);
  async_copy_b128(lds_addr32(&Bs[0][srow][sseg + 8]), gB + 8);

  const int nk = K / BK;
  for (int kt = 0; kt < nk; ++kt) {
    const int cur = kt & 1;
    wait_asynccnt0();          // our async writes into buf `cur` are done
    __syncthreads();           // everyone's are done; prior reads of `nxt` consumed

    if (kt + 1 < nk) {         // overlap next tile's DMA with this tile's WMMAs
      const int nxt = cur ^ 1;
      const unsigned short* pA = gA + (size_t)(kt + 1) * BK;
      const unsigned short* pB = gB + (size_t)(kt + 1) * BK;
      async_copy_b128(lds_addr32(&As[nxt][srow][sseg]),     pA);
      async_copy_b128(lds_addr32(&As[nxt][srow][sseg + 8]), pA + 8);
      async_copy_b128(lds_addr32(&Bs[nxt][srow][sseg]),     pB);
      async_copy_b128(lds_addr32(&Bs[nxt][srow][sseg + 8]), pB + 8);
    }

    BV a[4], b[2];
#pragma unroll
    for (int mi = 0; mi < 4; ++mi) {
      int row = wm + mi * 16 + r;
      a[mi].u[0] = *(const U128*)&As[cur][row][half * 8];
      a[mi].u[1] = *(const U128*)&As[cur][row][16 + half * 8];
    }
#pragma unroll
    for (int ni = 0; ni < 2; ++ni) {
      int col = wn + ni * 16 + r;
      b[ni].u[0] = *(const U128*)&Bs[cur][col][half * 8];
      b[ni].u[1] = *(const U128*)&Bs[cur][col][16 + half * 8];
    }
#pragma unroll
    for (int mi = 0; mi < 4; ++mi)
#pragma unroll
      for (int ni = 0; ni < 2; ++ni)
        acc[mi][ni] = wmma_bf16(a[mi].v, b[ni].v, acc[mi][ni]);
  }

#pragma unroll
  for (int mi = 0; mi < 4; ++mi)
#pragma unroll
    for (int ni = 0; ni < 2; ++ni)
#pragma unroll
      for (int i = 0; i < 8; ++i) {
        int row = m0 + wm + mi * 16 + ((lane < 16) ? i : i + 8);
        int col = n0 + wn + ni * 16 + r;
        C[(size_t)row * N + col] = acc[mi][ni][i] + bias[col];
      }
}

// ---------- kernel 2: RoPE + repack q,k -> [B,H,S,96] bf16 ; v -> [B,H,80,S] bf16 ----------
__global__ __launch_bounds__(256) void rope_repack(
    const float* __restrict__ qkv,   // [B*S, 3840] (bias applied)
    const float* __restrict__ cosb,  // [S, 80]
    const float* __restrict__ sinb,  // [S, 80]
    unsigned short* __restrict__ qb,
    unsigned short* __restrict__ kb,
    unsigned short* __restrict__ vt) {
  int blk = blockIdx.x;
  int b = blk >> 10;          // / S
  int s = blk & 1023;         // % S
  const float* row = qkv + (size_t)blk * D3;
  int tid = threadIdx.x;

  for (int j = tid; j < DD; j += 256) {
    int h = j / HD, hd = j - h * HD;
    float c  = cosb[s * HD + hd];
    float sn = sinb[s * HD + hd];
    int   j2  = h * HD + ((hd < 40) ? hd + 40 : hd - 40);
    float sgn = (hd < 40) ? -1.0f : 1.0f;
    float qv = row[j], kv = row[DD + j], vv = row[2 * DD + j];
    float qr = qv * c + sgn * row[j2] * sn;
    float kr = kv * c + sgn * row[DD + j2] * sn;
    size_t qi = (((size_t)(b * HH + h)) * SS + s) * HDP + hd;
    qb[qi] = f2bf(qr);
    kb[qi] = f2bf(kr);
    vt[(((size_t)(b * HH + h)) * HD + hd) * SS + s] = f2bf(vv);
  }
  // zero the K-padding columns 80..95 (they enter the WMMA K reduction)
  {
    int h = tid >> 4, hd = HD + (tid & 15);
    size_t qi = (((size_t)(b * HH + h)) * SS + s) * HDP + hd;
    qb[qi] = 0;
    kb[qi] = 0;
  }
}

// ---------- kernel 3: attention, one block per (b,h, 32-query tile) ----------
#define SP 1032   // padded LDS stride (elements) -> conflict-free b128 column reads

__global__ __launch_bounds__(256) void attn_kernel(
    const unsigned short* __restrict__ qb,   // [B,H,S,96]
    const unsigned short* __restrict__ kb,   // [B,H,S,96]
    const unsigned short* __restrict__ vt,   // [B,H,80,S]
    unsigned short* __restrict__ ctxb) {     // [B,S,D] bf16
  extern __shared__ char smem[];
  float*          sc  = (float*)smem;                                   // [32][SP] fp32
  unsigned short* pb  = (unsigned short*)(smem + 32 * SP * 4);          // [32][SP] bf16
  float*          red = (float*)(smem + 32 * SP * 4 + 32 * SP * 2);     // [32][8]

  const int blk  = blockIdx.x;
  const int qblk = blk & 31;        // S/32 tiles
  const int bh   = blk >> 5;        // b*16+h
  const int m0   = qblk * 32;
  const size_t qkbase = (size_t)bh * SS * HDP;
  const size_t vbase  = (size_t)bh * HD * SS;

  const int tid  = threadIdx.x;
  const int w    = tid >> 5;
  const int lane = tid & 31;
  const int r    = lane & 15;
  const int half = lane >> 4;

  // ---- phase 1: scores = (q @ k^T) * scale, full 32x1024 tile into LDS ----
  BV qop[2][3];
#pragma unroll
  for (int mi = 0; mi < 2; ++mi)
#pragma unroll
    for (int kc = 0; kc < 3; ++kc) {
      const unsigned short* p =
          qb + qkbase + (size_t)(m0 + mi * 16 + r) * HDP + kc * 32 + half * 8;
      qop[mi][kc].u[0] = *(const U128*)p;
      qop[mi][kc].u[1] = *(const U128*)(p + 16);
    }

  const float scale = 0.111803398874989485f;   // 1/sqrt(80)
  for (int kt = w; kt < 64; kt += 8) {         // 64 key tiles of 16
    BV bop[3];
#pragma unroll
    for (int kc = 0; kc < 3; ++kc) {
      const unsigned short* p =
          kb + qkbase + (size_t)(kt * 16 + r) * HDP + kc * 32 + half * 8;
      bop[kc].u[0] = *(const U128*)p;
      bop[kc].u[1] = *(const U128*)(p + 16);
    }
    v8f a0 = v8f_zero(), a1 = v8f_zero();
#pragma unroll
    for (int kc = 0; kc < 3; ++kc) {
      a0 = wmma_bf16(qop[0][kc].v, bop[kc].v, a0);
      a1 = wmma_bf16(qop[1][kc].v, bop[kc].v, a1);
    }
#pragma unroll
    for (int i = 0; i < 8; ++i) {
      int rr = (lane < 16) ? i : i + 8;
      int cc = kt * 16 + r;
      sc[rr * SP + cc]        = a0[i] * scale;
      sc[(16 + rr) * SP + cc] = a1[i] * scale;
    }
  }
  __syncthreads();

  // ---- phase 2: row softmax (8 threads per row, 128 cols each) ----
  {
    int srow = tid >> 3, sub = tid & 7;
    int c0 = sub * 128;
    float mx = -3.0e38f;
    for (int c = c0; c < c0 + 128; ++c) mx = fmaxf(mx, sc[srow * SP + c]);
    red[srow * 8 + sub] = mx;
    __syncthreads();
    mx = red[srow * 8];
#pragma unroll
    for (int t = 1; t < 8; ++t) mx = fmaxf(mx, red[srow * 8 + t]);
    __syncthreads();
    float sum = 0.0f;
    for (int c = c0; c < c0 + 128; ++c) {
      float e = __expf(sc[srow * SP + c] - mx);
      sc[srow * SP + c] = e;
      sum += e;
    }
    red[srow * 8 + sub] = sum;
    __syncthreads();
    sum = 0.0f;
#pragma unroll
    for (int t = 0; t < 8; ++t) sum += red[srow * 8 + t];
    float inv = 1.0f / sum;
    for (int c = c0; c < c0 + 128; ++c)
      pb[srow * SP + c] = f2bf(sc[srow * SP + c] * inv);
  }
  __syncthreads();

  // ---- phase 3: ctx = P[32x1024] @ V[1024x80] via WMMA ----
  const int h_ = bh & 15;
  const int b_ = bh >> 4;
  for (int task = w; task < 10; task += 8) {   // 2 m-tiles x 5 n-tiles
    int mi = task / 5;
    int ni = task % 5;
    v8f acc = v8f_zero();
    const int prow = mi * 16 + r;
    const unsigned short* vrow = vt + vbase + (size_t)(ni * 16 + r) * SS;
#pragma unroll 4
    for (int kk = 0; kk < 32; ++kk) {
      BV a, bb;
      const unsigned short* pa = pb + prow * SP + kk * 32 + half * 8;
      a.u[0] = *(const U128*)pa;
      a.u[1] = *(const U128*)(pa + 16);
      const unsigned short* pv = vrow + kk * 32 + half * 8;
      bb.u[0] = *(const U128*)pv;
      bb.u[1] = *(const U128*)(pv + 16);
      acc = wmma_bf16(a.v, bb.v, acc);
    }
#pragma unroll
    for (int i = 0; i < 8; ++i) {
      int s_out = m0 + mi * 16 + ((lane < 16) ? i : i + 8);
      int col   = h_ * HD + ni * 16 + r;
      ctxb[((size_t)(b_ * SS + s_out)) * DD + col] = f2bf(acc[i]);
    }
  }
}

// ---------- host side ----------
extern "C" void kernel_launch(void* const* d_in, const int* in_sizes, int n_in,
                              void* d_out, int out_size, void* d_ws, size_t ws_size,
                              hipStream_t stream) {
  (void)in_sizes; (void)n_in; (void)out_size; (void)ws_size;
  const float* x     = (const float*)d_in[0];
  const float* cosb  = (const float*)d_in[1];
  const float* sinb  = (const float*)d_in[2];
  const float* Wqkv  = (const float*)d_in[3];
  const float* bqkv  = (const float*)d_in[4];
  const float* Wproj = (const float*)d_in[5];
  const float* bproj = (const float*)d_in[6];
  float* out = (float*)d_out;

  char* ws = (char*)d_ws;
  size_t off = 0;
  auto carve = [&](size_t bytes) -> void* {
    void* p = ws + off;
    off += (bytes + 255) & ~(size_t)255;
    return p;
  };
  unsigned short* xb     = (unsigned short*)carve((size_t)BS * DD * 2);
  unsigned short* wqkvb  = (unsigned short*)carve((size_t)D3 * DD * 2);
  unsigned short* wprojb = (unsigned short*)carve((size_t)DD * DD * 2);
  float*          qkvf   = (float*)carve((size_t)BS * D3 * 4);
  unsigned short* qbuf   = (unsigned short*)carve((size_t)BB * HH * SS * HDP * 2);
  unsigned short* kbuf   = (unsigned short*)carve((size_t)BB * HH * SS * HDP * 2);
  unsigned short* vtb    = (unsigned short*)carve((size_t)BB * HH * HD * SS * 2);
  unsigned short* ctxb   = (unsigned short*)carve((size_t)BS * DD * 2);

  // 0) fp32 -> bf16 conversions
  {
    int n1 = BS * DD, n2 = D3 * DD, n3 = DD * DD;
    cvt_f32_bf16<<<(n1 + 255) / 256, 256, 0, stream>>>(x, xb, n1);
    cvt_f32_bf16<<<(n2 + 255) / 256, 256, 0, stream>>>(Wqkv, wqkvb, n2);
    cvt_f32_bf16<<<(n3 + 255) / 256, 256, 0, stream>>>(Wproj, wprojb, n3);
  }

  // 1) QKV GEMM: [4096,3840] = xb[4096,1280] @ Wqkv^T + bias
  {
    dim3 grid(D3 / BN, BS / BM);
    gemm_bf16_bias<<<grid, 256, 0, stream>>>(xb, wqkvb, bqkv, qkvf, BS, D3, DD);
  }

  // 2) RoPE + repack
  rope_repack<<<BB * SS, 256, 0, stream>>>(qkvf, cosb, sinb, qbuf, kbuf, vtb);

  // 3) attention
  {
    size_t smem = (size_t)32 * SP * 4 + (size_t)32 * SP * 2 + 32 * 8 * 4;
    (void)hipFuncSetAttribute((const void*)attn_kernel,
                              hipFuncAttributeMaxDynamicSharedMemorySize,
                              (int)smem);
    attn_kernel<<<BB * HH * (SS / 32), 256, smem, stream>>>(qbuf, kbuf, vtb, ctxb);
  }

  // 4) output projection: [4096,1280] = ctxb @ Wproj^T + bias
  {
    dim3 grid(DD / BN, BS / BM);
    gemm_bf16_bias<<<grid, 256, 0, stream>>>(ctxb, wprojb, bproj, out, BS, DD, DD);
  }
}
